// ConditionPredictorGAT_61211873902712
// MI455X (gfx1250) — matrix operand order repaired and checked by reading, not compiled, not run on hardware
//
#include <hip/hip_runtime.h>
#include <hip/hip_bf16.h>
#include <math.h>

// ---------------------------------------------------------------------------
// CDNA5 (gfx1250) GAT forward. Dense projections via v_wmma_f32_16x16x32_bf16
// (8 accumulators / wave; all fragment loads hoisted ahead of the WMMA burst
// so loads overlap the matrix pipe), edge softmax via f32 atomics. wave32.
// ---------------------------------------------------------------------------

typedef __attribute__((ext_vector_type(16))) __bf16 v16bf;
typedef __attribute__((ext_vector_type(8)))  __bf16 v8bf;
typedef __attribute__((ext_vector_type(8)))  float  v8f;

#define HID 128

// ---------------- generic helpers ----------------
__global__ void k_zero_f32(float* p, long long n) {
  long long i = (long long)blockIdx.x * blockDim.x + threadIdx.x;
  if (i < n) p[i] = 0.f;
}
__global__ void k_zero_u32(unsigned* p, long long n) {
  long long i = (long long)blockIdx.x * blockDim.x + threadIdx.x;
  if (i < n) p[i] = 0u;   // 0 < encode(x) for every float x -> valid -inf key
}
// float4-wide f32 -> bf16 convert (n must be a multiple of 4: N*fi always is)
__global__ void k_cvt_bf16(const float* __restrict__ s, __bf16* __restrict__ d,
                           long long n) {
  long long i = ((long long)blockIdx.x * blockDim.x + threadIdx.x) * 4;
  if (i >= n) return;
  float4 v = *(const float4*)(s + i);
  d[i + 0] = (__bf16)v.x; d[i + 1] = (__bf16)v.y;
  d[i + 2] = (__bf16)v.z; d[i + 3] = (__bf16)v.w;
}

__device__ __forceinline__ unsigned enc_f32(float f) {
  unsigned b = __float_as_uint(f);
  return (b & 0x80000000u) ? ~b : (b | 0x80000000u);
}
__device__ __forceinline__ float dec_f32(unsigned k) {
  unsigned b = (k & 0x80000000u) ? (k & 0x7FFFFFFFu) : ~k;
  return __uint_as_float(b);
}

// ---------------- edge attr + mean ----------------
__global__ void k_eattr(const float* __restrict__ x, const int* __restrict__ ei,
                        const float* __restrict__ ew, const int* __restrict__ code_table,
                        const int* n_tl, const int* n_med,
                        float* __restrict__ eattr, float* __restrict__ sum_ws,
                        int E, int NCODES, int NC) {
  __shared__ float red[256];
  int e = blockIdx.x * blockDim.x + threadIdx.x;
  float v = 0.f;
  if (e < E) {
    int d = ei[E + e];
    bool mask = d >= (*n_tl + *n_med);
    int code = (int)x[(size_t)d * HID];           // x[:,0] holds integral codes
    code = code < 0 ? 0 : (code > NCODES - 1 ? NCODES - 1 : code);
    int idx = code_table[code];
    int ci  = idx > 0 ? idx : 0;
    ci = ci < NC ? ci : NC - 1;
    float w = ew[ci];
    v = (mask && idx >= 0) ? w : 1.0f;
    eattr[e] = v;
  }
  red[threadIdx.x] = v;
  __syncthreads();
  for (int s = 128; s > 0; s >>= 1) {
    if (threadIdx.x < s) red[threadIdx.x] += red[threadIdx.x + s];
    __syncthreads();
  }
  if (threadIdx.x == 0) atomicAdd(sum_ws, red[0]);
}
__global__ void k_mean(const float* sum_ws, float* mean_ws, int E) {
  *mean_ws = *sum_ws / (float)E;
}

// ---------------- weight packing for WMMA B fragments ----------------
// B fragment of v_wmma_f32_16x16x32_bf16: lane l holds column n=l&15,
// K = (l>>4)*8 + i (i<8), 16 + (l>>4)*8 + i-8 (i>=8), packed 2 per VGPR -> store 16 contiguous bf16
// per lane per (ktile,ntile) so the GEMM loads one 32B vector per fragment.
__global__ void k_pack_w(const float* __restrict__ W, __bf16* __restrict__ Bp,
                         int fi, int fo) {
  int nT = fo >> 4, kT = fi >> 5;
  int t = blockIdx.x * blockDim.x + threadIdx.x;     // one (tile,lane) each
  int total = kT * nT * 32;
  if (t >= total) return;
  int lane = t & 31, tile = t >> 5;
  int nt = tile % nT, kt = tile / nT;
  int n = lane & 15;
  int kbase = kt * 32 + (lane >> 4) * 8;
  __bf16* dst = Bp + (size_t)t * 16;
#pragma unroll
  for (int i = 0; i < 16; ++i)
    dst[i] = (__bf16)W[(size_t)(kbase + (i < 8 ? i : i + 8)) * fo + nt * 16 + n];
}

// ---------------- WMMA GEMM: C[N,fo] = A[N,fi](bf16) * B[fi,fo](packed) ----
// Each wave computes a 16(M) x 128(N) strip: one A fragment feeds 8 WMMAs.
// All 8 B fragments + A are loaded BEFORE the WMMA burst so the compiler can
// clause the loads and drain them with a descending s_wait_loadcnt watermark,
// overlapping memory with the matrix pipe.
// grid.x = ceil(N/64), grid.y = fo/128, block = 128 (4 waves, 1 mtile/wave).
__global__ void k_gemm_wmma(const __bf16* __restrict__ A, const __bf16* __restrict__ Bp,
                            float* __restrict__ C, int N, int fi, int fo) {
  int lane  = threadIdx.x & 31;
  int wave  = threadIdx.x >> 5;
  int mtile = blockIdx.x * 4 + wave;
  int nbase = blockIdx.y * 8;                        // first of 8 ntiles
  int nT = fo >> 4, kT = fi >> 5;
  int half = lane >> 4;
  int mloc = lane & 15;
  int row  = mtile * 16 + mloc;
  int arow = row < N ? row : 0;                      // clamp; clamped rows never stored
  const __bf16* ap = A + (size_t)arow * fi + half * 8;
  v8f c[8] = {};
  for (int kt = 0; kt < kT; ++kt) {
    if (kt + 2 < kT)                                 // global_prefetch_b8 on A stream
      __builtin_prefetch((const void*)(ap + (kt + 2) * 32), 0, 1);
    // ---- issue ALL loads for this k-step first ----
    const __bf16* ak = ap + kt * 32;
    v8bf a0 = *(const v8bf*)(ak);
    v8bf a1 = *(const v8bf*)(ak + 16);
    const __bf16* bp = Bp + (((size_t)kt * nT + nbase) * 32 + lane) * 16;
    v16bf b[8];
#pragma unroll
    for (int j = 0; j < 8; ++j)
      b[j] = *(const v16bf*)(bp + (size_t)j * 32 * 16);
    // A fragment: lane (half,m) needs K = half*8+0..7 and 16+half*8+0..7
    v16bf a;
#pragma unroll
    for (int i = 0; i < 8; ++i) { a[i] = a0[i]; a[i + 8] = a1[i]; }
    // ---- WMMA burst ----
#pragma unroll
    for (int j = 0; j < 8; ++j)
      c[j] = __builtin_amdgcn_wmma_f32_16x16x32_bf16(false, a, false, b[j], (short)0,
                                                     c[j], false, false);
  }
#pragma unroll
  for (int j = 0; j < 8; ++j) {
    int col = (nbase + j) * 16 + mloc;
#pragma unroll
    for (int v = 0; v < 8; ++v) {
      int r = mtile * 16 + v + 8 * half;             // C layout: VGPR v -> M = v + 8*half
      if (r < N) C[(size_t)r * fo + col] = c[j][v];
    }
  }
}

// ---------------- attention scalar projections ----------------
// a_s[n,h] = <xl[n,h,:], att_src[h,:]>, one wave per (n,h)
__global__ void k_att_dots(const float* __restrict__ xl, const float* __restrict__ att_s,
                           const float* __restrict__ att_d, float* __restrict__ a_s,
                           float* __restrict__ a_d, int N, int H) {
  int lane = threadIdx.x & 31;
  int idx = blockIdx.x * (blockDim.x >> 5) + (threadIdx.x >> 5);
  if (idx >= N * H) return;
  int n = idx / H, h = idx - n * H;
  const float* xp = xl + ((size_t)n * H + h) * HID;
  const float* sp = att_s + h * HID;
  const float* dp = att_d + h * HID;
  float ss = 0.f, sd = 0.f;
  for (int i = lane; i < HID; i += 32) { float v = xp[i]; ss += v * sp[i]; sd += v * dp[i]; }
#pragma unroll
  for (int off = 16; off; off >>= 1) { ss += __shfl_xor(ss, off, 32); sd += __shfl_xor(sd, off, 32); }
  if (lane == 0) { a_s[idx] = ss; a_d[idx] = sd; }
}

// s_e[h] = <W_e[0, h*128:...], att_e[h,:]>  (edge term collapses to a scalar/head)
__global__ void k_se(const float* __restrict__ We, const float* __restrict__ att_e,
                     float* __restrict__ s_e, int H) {
  int lane = threadIdx.x & 31;
  int h = threadIdx.x >> 5;
  if (h >= H) return;
  float s = 0.f;
  for (int i = lane; i < HID; i += 32) s += We[h * HID + i] * att_e[h * HID + i];
#pragma unroll
  for (int off = 16; off; off >>= 1) s += __shfl_xor(s, off, 32);
  if (lane == 0) s_e[h] = s;
}

// ---------------- edge softmax passes ----------------
__global__ void k_alpha_max(const int* __restrict__ ei, const float* __restrict__ eattr,
                            const float* __restrict__ emean, const float* __restrict__ a_s,
                            const float* __restrict__ a_d, const float* __restrict__ s_e,
                            float* __restrict__ alpha, unsigned* __restrict__ mseg,
                            int E, int E2, int H) {
  int e = blockIdx.x * blockDim.x + threadIdx.x;
  if (e >= E2) return;
  int s, d; float ea;
  if (e < E) { s = ei[e]; d = ei[E + e]; ea = eattr[e]; }
  else       { s = e - E; d = e - E;     ea = *emean;  }
  for (int h = 0; h < H; ++h) {
    float al = a_s[s * H + h] + a_d[d * H + h] + ea * s_e[h];
    al = al > 0.f ? al : 0.2f * al;                  // leaky_relu(0.2)
    alpha[(size_t)e * H + h] = al;
    atomicMax(&mseg[d * H + h], enc_f32(al));
  }
}

__global__ void k_exp_den(const int* __restrict__ ei, float* __restrict__ alpha,
                          const unsigned* __restrict__ mseg, float* __restrict__ den,
                          int E, int E2, int H) {
  int e = blockIdx.x * blockDim.x + threadIdx.x;
  if (e >= E2) return;
  int d = (e < E) ? ei[E + e] : e - E;
  for (int h = 0; h < H; ++h) {
    float ex = expf(alpha[(size_t)e * H + h] - dec_f32(mseg[d * H + h]));
    alpha[(size_t)e * H + h] = ex;                   // overwrite with numerator
    atomicAdd(&den[d * H + h], ex);
  }
}

// acc[dst, :] += (ex/den[dst]) * xl[src, :]   -- one block per edge, float4 wide
__global__ void k_scatter(const int* __restrict__ ei, const float* __restrict__ ex,
                          const float* __restrict__ den, const float* __restrict__ xl,
                          float* __restrict__ acc, int E, int H, int fo) {
  int e = blockIdx.x;
  int s, d;
  if (e < E) { s = ei[e]; d = ei[E + e]; } else { s = e - E; d = e - E; }
  for (int c = threadIdx.x * 4; c < fo; c += blockDim.x * 4) {
    int h = c >> 7;                                  // HID == 128, 4-chunk stays in-head
    float w = ex[(size_t)e * H + h] / den[d * H + h];
    float4 v = *(const float4*)(xl + (size_t)s * fo + c);
    float* ap = acc + (size_t)d * fo + c;
    atomicAdd(ap + 0, w * v.x);
    atomicAdd(ap + 1, w * v.y);
    atomicAdd(ap + 2, w * v.z);
    atomicAdd(ap + 3, w * v.w);
  }
}

__global__ void k_bias_elu(float* __restrict__ acc, const float* __restrict__ b,
                           int N, int fo) {
  long long i = (long long)blockIdx.x * blockDim.x + threadIdx.x;
  if (i >= (long long)N * fo) return;
  float v = acc[i] + b[i % fo];
  acc[i] = v > 0.f ? v : expm1f(v);                  // elu, alpha=1
}

// ---------------- head MLPs ----------------
__global__ void k_comb(const float* __restrict__ h3, const int* __restrict__ last,
                       const float* __restrict__ ew, float* __restrict__ comb, int NC) {
  int t = blockIdx.x * blockDim.x + threadIdx.x;
  if (t < HID) comb[t] = h3[(size_t)(*last) * HID + t];
  else if (t < HID + NC) comb[t] = ew[t - HID];
}
__global__ void k_head_z(const float* __restrict__ comb, const float* __restrict__ W1,
                         const float* __restrict__ b1, float* __restrict__ z, int CIN) {
  int j = threadIdx.x;                               // 128 threads
  float acc = b1[j];
  for (int i = 0; i < CIN; ++i) acc += comb[i] * W1[(size_t)i * HID + j];
  z[j] = acc > 0.f ? acc : 0.f;
}
__global__ void k_head_o(const float* __restrict__ z, const float* __restrict__ W2,
                         const float* __restrict__ b2, float* __restrict__ prob, int NC) {
  int o = blockIdx.x * blockDim.x + threadIdx.x;
  if (o >= NC) return;
  float acc = b2[o];
  for (int j = 0; j < HID; ++j) acc += z[j] * W2[(size_t)j * NC + o];
  prob[o] = 1.f / (1.f + expf(-acc));
}
__global__ void k_final(const float* __restrict__ startp, const float* __restrict__ endp,
                        const float* __restrict__ ew, float* __restrict__ out, int NC) {
  int i = blockIdx.x * blockDim.x + threadIdx.x;
  if (i >= NC) return;
  float w = ew[i];
  float prob = (w > 0.f) ? (1.f - endp[i]) : startp[i];
  float pred = rintf(prob);                          // round half to even, like jnp.round
  out[i] = pred;
  out[NC + i] = pred - w;
}

// ---------------------------------------------------------------------------
extern "C" void kernel_launch(void* const* d_in, const int* in_sizes, int n_in,
                              void* d_out, int out_size, void* d_ws, size_t ws_size,
                              hipStream_t stream) {
  // ---- inputs (setup_inputs insertion order; nested dicts in insertion order)
  const float* x   = (const float*)d_in[0];
  const int*   ei  = (const int*)d_in[1];
  const float* ew  = (const float*)d_in[2];
  const int*   last = (const int*)d_in[3];
  const int*   n_tl = (const int*)d_in[4];
  const int*   n_med = (const int*)d_in[5];
  const int*   code_table = (const int*)d_in[6];
  // params: g1,g2,g3 each {W, att_src, att_dst, W_e, att_e, b}
  const float* g_W[3]   = {(const float*)d_in[7],  (const float*)d_in[13], (const float*)d_in[19]};
  const float* g_as[3]  = {(const float*)d_in[8],  (const float*)d_in[14], (const float*)d_in[20]};
  const float* g_ad[3]  = {(const float*)d_in[9],  (const float*)d_in[15], (const float*)d_in[21]};
  const float* g_We[3]  = {(const float*)d_in[10], (const float*)d_in[16], (const float*)d_in[22]};
  const float* g_ae[3]  = {(const float*)d_in[11], (const float*)d_in[17], (const float*)d_in[23]};
  const float* g_b[3]   = {(const float*)d_in[12], (const float*)d_in[18], (const float*)d_in[24]};
  const float* s_W1 = (const float*)d_in[25]; const float* s_b1 = (const float*)d_in[26];
  const float* s_W2 = (const float*)d_in[27]; const float* s_b2 = (const float*)d_in[28];
  const float* e_W1 = (const float*)d_in[29]; const float* e_b1 = (const float*)d_in[30];
  const float* e_W2 = (const float*)d_in[31]; const float* e_b2 = (const float*)d_in[32];

  const int N  = in_sizes[0] / HID;        // 50000
  const int E  = in_sizes[1] / 2;          // 500000
  const int NC = in_sizes[2];              // 1024
  const int NCODES = in_sizes[6];          // 512
  const int E2 = E + N;
  const int HMAX = 4, FMAX = 512;

  // ---- workspace carve-up
  char* ws = (char*)d_ws;
  size_t off = 0;
  auto alloc = [&](size_t bytes) { char* p = ws + off; off += (bytes + 255) & ~(size_t)255; return (void*)p; };
  float*  bufA  = (float*)alloc((size_t)N * FMAX * 4);       // xl (projection)
  float*  bufB  = (float*)alloc((size_t)N * FMAX * 4);       // acc / layer output
  __bf16* xbf   = (__bf16*)alloc((size_t)N * FMAX * 2);      // bf16 activations
  __bf16* packB = (__bf16*)alloc(512 * 1024);                // packed weights
  float*  eattr = (float*)alloc((size_t)E * 4);
  float*  alpha = (float*)alloc((size_t)E2 * HMAX * 4);
  float*  a_s   = (float*)alloc((size_t)N * HMAX * 4);
  float*  a_d   = (float*)alloc((size_t)N * HMAX * 4);
  unsigned* mseg = (unsigned*)alloc((size_t)N * HMAX * 4);
  float*  den   = (float*)alloc((size_t)N * HMAX * 4);
  float*  s_e   = (float*)alloc(64);
  float*  msum  = (float*)alloc(64);       // [0]=sum, mean kernel writes [1]
  float*  emean = msum + 1;
  float*  comb  = (float*)alloc((HID + NC) * 4);
  float*  zbuf  = (float*)alloc(2 * HID * 4);
  float*  prob  = (float*)alloc(2 * NC * 4);
  (void)ws_size; (void)n_in; (void)out_size;

  const int B256 = 256;
  auto cdiv = [](long long a, long long b) { return (int)((a + b - 1) / b); };

  // ---- edge attributes + mean
  k_zero_f32<<<1, 32, 0, stream>>>(msum, 2);
  k_eattr<<<cdiv(E, B256), B256, 0, stream>>>(x, ei, ew, code_table, n_tl, n_med,
                                              eattr, msum, E, NCODES, NC);
  k_mean<<<1, 1, 0, stream>>>(msum, emean, E);

  // ---- three GAT layers
  const int fis[3]   = {HID, 4 * HID, 2 * HID};
  const int heads[3] = {4, 2, 1};
  const float* input = x;
  for (int L = 0; L < 3; ++L) {
    int fi = fis[L], H = heads[L], fo = H * HID;
    long long nin = (long long)N * fi, nout = (long long)N * fo;

    // bf16 activations + packed weights
    k_cvt_bf16<<<cdiv(nin / 4, B256), B256, 0, stream>>>(input, xbf, nin);
    int tiles = (fi >> 5) * (fo >> 4) * 32;
    k_pack_w<<<cdiv(tiles, B256), B256, 0, stream>>>(g_W[L], packB, fi, fo);

    // xl = x @ W via WMMA (bufA); wave = 16x128 strip, 8 accumulators
    dim3 gg(cdiv(N, 64), fo / 128);
    k_gemm_wmma<<<gg, 128, 0, stream>>>(xbf, packB, bufA, N, fi, fo);

    // attention scalars
    k_att_dots<<<cdiv((long long)N * H, 4), 128, 0, stream>>>(bufA, g_as[L], g_ad[L],
                                                              a_s, a_d, N, H);
    k_se<<<1, 128, 0, stream>>>(g_We[L], g_ae[L], s_e, H);

    // segment softmax over edges
    k_zero_u32<<<cdiv((long long)N * H, B256), B256, 0, stream>>>(mseg, (long long)N * H);
    k_zero_f32<<<cdiv((long long)N * H, B256), B256, 0, stream>>>(den, (long long)N * H);
    k_zero_f32<<<cdiv(nout, B256), B256, 0, stream>>>(bufB, nout);
    k_alpha_max<<<cdiv(E2, B256), B256, 0, stream>>>(ei, eattr, emean, a_s, a_d, s_e,
                                                     alpha, mseg, E, E2, H);
    k_exp_den<<<cdiv(E2, B256), B256, 0, stream>>>(ei, alpha, mseg, den, E, E2, H);
    k_scatter<<<E2, 128, 0, stream>>>(ei, alpha, den, bufA, bufB, E, H, fo);
    k_bias_elu<<<cdiv(nout, B256), B256, 0, stream>>>(bufB, g_b[L], N, fo);
    input = bufB;                          // next layer reads bufB (freed after cvt)
  }

  // ---- heads + final output
  k_comb<<<cdiv(HID + NC, B256), B256, 0, stream>>>(bufB, last, ew, comb, NC);
  k_head_z<<<1, HID, 0, stream>>>(comb, s_W1, s_b1, zbuf,       HID + NC);
  k_head_z<<<1, HID, 0, stream>>>(comb, e_W1, e_b1, zbuf + HID, HID + NC);
  k_head_o<<<cdiv(NC, B256), B256, 0, stream>>>(zbuf,       s_W2, s_b2, prob,      NC);
  k_head_o<<<cdiv(NC, B256), B256, 0, stream>>>(zbuf + HID, e_W2, e_b2, prob + NC, NC);
  k_final<<<cdiv(NC, B256), B256, 0, stream>>>(prob, prob + NC, ew, (float*)d_out, NC);
}
